// AttensionBlock_79027398246589
// MI455X (gfx1250) — matrix operand order repaired and checked by reading, not compiled, not run on hardware
//
#include <hip/hip_runtime.h>
#include <math.h>

#define C_DIM  512
#define B_DIM  16
#define HW_DIM 1024
#define G_DIM  32
#define CPG    16          // channels per group
#define GN_EPS 1e-5f
#define GRP    4           // batches processed per launch group

typedef __bf16 bf16_t;
typedef __attribute__((ext_vector_type(16))) __bf16 v16bf;
typedef __attribute__((ext_vector_type(8)))  float  v8f;

union FragU { v16bf v; uint4 q[2]; };

// A-matrix 16x32 bf16 fragment (this code's k order, same as B): lane holds row m=lane%16;
// lanes 0-15: K = k0+0..15 ; lanes 16-31: K = k0+16..31 (contiguous).
__device__ __forceinline__ v16bf load_frag_a(const bf16_t* p, int hi) {
  FragU f;
  const bf16_t* pp = p + hi * 16;
  f.q[0] = *reinterpret_cast<const uint4*>(pp);        // K  +0..7
  f.q[1] = *reinterpret_cast<const uint4*>(pp + 8);    // K  +8..15
  return f.v;
}

// B-matrix 32x16 bf16 fragment: lane holds col n=lane%16;
// lanes 0-15: K = k0+0..15 ; lanes 16-31: K = k0+16..31 (contiguous).
__device__ __forceinline__ v16bf load_frag_b(const bf16_t* p, int hi) {
  FragU f;
  const bf16_t* pp = p + hi * 16;
  f.q[0] = *reinterpret_cast<const uint4*>(pp);
  f.q[1] = *reinterpret_cast<const uint4*>(pp + 8);
  return f.v;
}

// One wave computes a 32x64 tile of an NT GEMM:
// D[m,n] = sum_k A[m*lda+k] * B[n*ldb+k]   (both operands K-contiguous)
// 8 WMMAs per k-step for 6 fragment fetches -> ~22 flop / L2 byte.
__device__ __forceinline__ void wave_gemm_32x64(const bf16_t* __restrict__ A, int lda,
                                                const bf16_t* __restrict__ B, int ldb,
                                                int K, int m0, int n0, v8f acc[2][4]) {
  const int lane = threadIdx.x & 31;
  const int lrow = lane & 15;
  const int hi   = lane >> 4;
  const bf16_t* ap[2];
  const bf16_t* bp[4];
  ap[0] = A + (size_t)(m0 +      lrow) * lda;
  ap[1] = A + (size_t)(m0 + 16 + lrow) * lda;
  bp[0] = B + (size_t)(n0 +      lrow) * ldb;
  bp[1] = B + (size_t)(n0 + 16 + lrow) * ldb;
  bp[2] = B + (size_t)(n0 + 32 + lrow) * ldb;
  bp[3] = B + (size_t)(n0 + 48 + lrow) * ldb;
  for (int k0 = 0; k0 < K; k0 += 32) {
    v16bf a0 = load_frag_a(ap[0] + k0, hi);
    v16bf a1 = load_frag_a(ap[1] + k0, hi);
    v16bf b0 = load_frag_b(bp[0] + k0, hi);
    v16bf b1 = load_frag_b(bp[1] + k0, hi);
    v16bf b2 = load_frag_b(bp[2] + k0, hi);
    v16bf b3 = load_frag_b(bp[3] + k0, hi);
    acc[0][0] = __builtin_amdgcn_wmma_f32_16x16x32_bf16(false, a0, false, b0, (short)0, acc[0][0], false, false);
    acc[0][1] = __builtin_amdgcn_wmma_f32_16x16x32_bf16(false, a0, false, b1, (short)0, acc[0][1], false, false);
    acc[0][2] = __builtin_amdgcn_wmma_f32_16x16x32_bf16(false, a0, false, b2, (short)0, acc[0][2], false, false);
    acc[0][3] = __builtin_amdgcn_wmma_f32_16x16x32_bf16(false, a0, false, b3, (short)0, acc[0][3], false, false);
    acc[1][0] = __builtin_amdgcn_wmma_f32_16x16x32_bf16(false, a1, false, b0, (short)0, acc[1][0], false, false);
    acc[1][1] = __builtin_amdgcn_wmma_f32_16x16x32_bf16(false, a1, false, b1, (short)0, acc[1][1], false, false);
    acc[1][2] = __builtin_amdgcn_wmma_f32_16x16x32_bf16(false, a1, false, b2, (short)0, acc[1][2], false, false);
    acc[1][3] = __builtin_amdgcn_wmma_f32_16x16x32_bf16(false, a1, false, b3, (short)0, acc[1][3], false, false);
  }
}

// ---------------------------------------------------------------- weights f32 -> bf16
__global__ void __launch_bounds__(256) wconv_kernel(const float* __restrict__ Wq,
                                                    const float* __restrict__ Wk,
                                                    const float* __restrict__ Wv,
                                                    const float* __restrict__ Wo,
                                                    bf16_t* __restrict__ Wall) {
  const int which = blockIdx.y;
  const float* src = (which == 0) ? Wq : (which == 1) ? Wk : (which == 2) ? Wv : Wo;
  bf16_t* dst = Wall + (size_t)which * C_DIM * C_DIM;
  const int i = blockIdx.x * 256 + threadIdx.x;
  if (i < C_DIM * C_DIM) dst[i] = (bf16_t)src[i];
}

// ------------------------------------------- GroupNorm + SiLU -> Xt[b][n][c] (bf16)
__global__ void __launch_bounds__(256) gn_silu_kernel(const float* __restrict__ x,
                                                      const float* __restrict__ gamma,
                                                      const float* __restrict__ beta,
                                                      bf16_t* __restrict__ Xt) {
  const int b = blockIdx.x / G_DIM;
  const int g = blockIdx.x % G_DIM;
  const int tid = threadIdx.x;
  const float* xp = x + ((size_t)b * C_DIM + (size_t)g * CPG) * HW_DIM;
  __shared__ float s1[256], s2[256];
  float sum = 0.f, sq = 0.f;
  for (int i = tid; i < CPG * HW_DIM; i += 256) { float v = xp[i]; sum += v; sq += v * v; }
  s1[tid] = sum; s2[tid] = sq; __syncthreads();
  for (int off = 128; off > 0; off >>= 1) {
    if (tid < off) { s1[tid] += s1[tid + off]; s2[tid] += s2[tid + off]; }
    __syncthreads();
  }
  const float invn = 1.f / (float)(CPG * HW_DIM);
  const float mean = s1[0] * invn;
  const float var  = s2[0] * invn - mean * mean;
  const float rstd = rsqrtf(var + GN_EPS);
  bf16_t* xo = Xt + (size_t)b * HW_DIM * C_DIM;
  for (int i = tid; i < CPG * HW_DIM; i += 256) {
    const int cl = i / HW_DIM, n = i % HW_DIM, c = g * CPG + cl;
    float v = (xp[i] - mean) * rstd;
    v = v * gamma[c] + beta[c];
    v = v / (1.f + __expf(-v));                 // SiLU
    xo[(size_t)n * C_DIM + c] = (bf16_t)v;
  }
}

// -------------------------------------- Q/K/V projection (blockIdx.y selects which,
//                                        blockIdx.z = batch within 4-batch group)
__global__ void __launch_bounds__(256) qkv_gemm_kernel(const bf16_t* __restrict__ Wall,
                                                       const float* __restrict__ bq,
                                                       const float* __restrict__ bk,
                                                       const float* __restrict__ bv,
                                                       const bf16_t* __restrict__ XtBase,
                                                       int b0,
                                                       bf16_t* __restrict__ QtBase,
                                                       bf16_t* __restrict__ KtBase,
                                                       bf16_t* __restrict__ VmBase,
                                                       float qscale) {
  const int z = blockIdx.z;
  const bf16_t* Xt = XtBase + (size_t)(b0 + z) * HW_DIM * C_DIM;
  bf16_t* Qt = QtBase + (size_t)z * HW_DIM * C_DIM;
  bf16_t* Kt = KtBase + (size_t)z * HW_DIM * C_DIM;
  bf16_t* Vm = VmBase + (size_t)z * HW_DIM * C_DIM;
  const int which = blockIdx.y;
  const bf16_t* W = Wall + (size_t)which * C_DIM * C_DIM;
  const float* bias = (which == 0) ? bq : (which == 1) ? bk : bv;
  const float scale = (which == 0) ? qscale : 1.0f;
  const int wave = (blockIdx.x * blockDim.x + threadIdx.x) >> 5;
  const int tiles_n = HW_DIM / 64;              // 16
  const int m0 = (wave / tiles_n) * 32;         // output channel o
  const int n0 = (wave % tiles_n) * 64;         // pixel n
  v8f acc[2][4] = {};
  wave_gemm_32x64(W, C_DIM, Xt, C_DIM, C_DIM, m0, n0, acc);
  const int lane = threadIdx.x & 31, lrow = lane & 15, hi = lane >> 4;
  for (int i = 0; i < 2; ++i)
    for (int j = 0; j < 4; ++j)
      for (int r = 0; r < 8; ++r) {
        const int o = m0 + i * 16 + r + hi * 8;
        const int n = n0 + j * 16 + lrow;
        const float v = (acc[i][j][r] + bias[o]) * scale;
        if (which < 2) {                                   // Q,K stored [n][c]
          bf16_t* O = (which == 0) ? Qt : Kt;
          O[(size_t)n * C_DIM + o] = (bf16_t)v;
        } else {                                           // V stored [c][n]
          Vm[(size_t)o * HW_DIM + n] = (bf16_t)v;
        }
      }
}

// ----------------------------------------------- S = (scale*Q)^T K  -> f32 [nq][nk]
__global__ void __launch_bounds__(256) scores_gemm_kernel(const bf16_t* __restrict__ QtBase,
                                                          const bf16_t* __restrict__ KtBase,
                                                          float* __restrict__ SBase) {
  const int z = blockIdx.z;
  const bf16_t* Qt = QtBase + (size_t)z * HW_DIM * C_DIM;
  const bf16_t* Kt = KtBase + (size_t)z * HW_DIM * C_DIM;
  float* S = SBase + (size_t)z * HW_DIM * HW_DIM;
  const int wave = (blockIdx.x * blockDim.x + threadIdx.x) >> 5;
  const int tiles_n = HW_DIM / 64;              // 16
  const int m0 = (wave / tiles_n) * 32;
  const int n0 = (wave % tiles_n) * 64;
  v8f acc[2][4] = {};
  wave_gemm_32x64(Qt, C_DIM, Kt, C_DIM, C_DIM, m0, n0, acc);
  const int lane = threadIdx.x & 31, lrow = lane & 15, hi = lane >> 4;
  for (int i = 0; i < 2; ++i)
    for (int j = 0; j < 4; ++j)
      for (int r = 0; r < 8; ++r)
        S[(size_t)(m0 + i * 16 + r + hi * 8) * HW_DIM + (n0 + j * 16 + lrow)] = acc[i][j][r];
}

// ------------------------------------------------ row softmax, f32 in / bf16 attn out
__global__ void __launch_bounds__(256) softmax_kernel(const float* __restrict__ SBase,
                                                      bf16_t* __restrict__ ABase) {
  __shared__ float red[256];
  const int z = blockIdx.y;
  const float* S = SBase + (size_t)z * HW_DIM * HW_DIM;
  bf16_t* A = ABase + (size_t)z * HW_DIM * HW_DIM;
  const int row = blockIdx.x, tid = threadIdx.x;
  const float* s = S + (size_t)row * HW_DIM;
  float lv[4];
  float mx = -3.4e38f;
  for (int t = 0; t < 4; ++t) { lv[t] = s[tid + t * 256]; mx = fmaxf(mx, lv[t]); }
  red[tid] = mx; __syncthreads();
  for (int off = 128; off > 0; off >>= 1) {
    if (tid < off) red[tid] = fmaxf(red[tid], red[tid + off]);
    __syncthreads();
  }
  mx = red[0]; __syncthreads();
  float sum = 0.f;
  for (int t = 0; t < 4; ++t) { lv[t] = __expf(lv[t] - mx); sum += lv[t]; }
  red[tid] = sum; __syncthreads();
  for (int off = 128; off > 0; off >>= 1) {
    if (tid < off) red[tid] += red[tid + off];
    __syncthreads();
  }
  const float inv = 1.f / red[0];
  bf16_t* a = A + (size_t)row * HW_DIM;
  for (int t = 0; t < 4; ++t) a[tid + t * 256] = (bf16_t)(lv[t] * inv);
}

// ------------------------------------------- H[n][c] = sum_m attn[n][m] * V[c][m]
__global__ void __launch_bounds__(256) av_gemm_kernel(const bf16_t* __restrict__ AttnBase,
                                                      const bf16_t* __restrict__ VmBase,
                                                      bf16_t* __restrict__ HbBase) {
  const int z = blockIdx.z;
  const bf16_t* Attn = AttnBase + (size_t)z * HW_DIM * HW_DIM;
  const bf16_t* Vm = VmBase + (size_t)z * HW_DIM * C_DIM;
  bf16_t* Hb = HbBase + (size_t)z * HW_DIM * C_DIM;
  const int wave = (blockIdx.x * blockDim.x + threadIdx.x) >> 5;
  const int tiles_n = C_DIM / 64;               // 8
  const int m0 = (wave / tiles_n) * 32;         // query pixel n
  const int n0 = (wave % tiles_n) * 64;         // channel c
  v8f acc[2][4] = {};
  wave_gemm_32x64(Attn, HW_DIM, Vm, HW_DIM, HW_DIM, m0, n0, acc);
  const int lane = threadIdx.x & 31, lrow = lane & 15, hi = lane >> 4;
  for (int i = 0; i < 2; ++i)
    for (int j = 0; j < 4; ++j)
      for (int r = 0; r < 8; ++r)
        Hb[(size_t)(m0 + i * 16 + r + hi * 8) * C_DIM + (n0 + j * 16 + lrow)] =
            (bf16_t)acc[i][j][r];
}

// --------------------------- out[o][n] = Wo*H + bo + x  (f32 stores to d_out slice)
__global__ void __launch_bounds__(256) out_gemm_kernel(const bf16_t* __restrict__ Wo,
                                                       const float* __restrict__ bo,
                                                       const bf16_t* __restrict__ HbBase,
                                                       const float* __restrict__ xBase,
                                                       float* __restrict__ OutBase,
                                                       int b0) {
  const int z = blockIdx.z;
  const bf16_t* Hb = HbBase + (size_t)z * HW_DIM * C_DIM;
  const float* xres = xBase + (size_t)(b0 + z) * C_DIM * HW_DIM;
  float* Out = OutBase + (size_t)(b0 + z) * C_DIM * HW_DIM;
  const int wave = (blockIdx.x * blockDim.x + threadIdx.x) >> 5;
  const int tiles_n = HW_DIM / 64;              // 16
  const int m0 = (wave / tiles_n) * 32;         // channel o
  const int n0 = (wave % tiles_n) * 64;         // pixel n
  v8f acc[2][4] = {};
  wave_gemm_32x64(Wo, C_DIM, Hb, C_DIM, C_DIM, m0, n0, acc);
  const int lane = threadIdx.x & 31, lrow = lane & 15, hi = lane >> 4;
  for (int i = 0; i < 2; ++i)
    for (int j = 0; j < 4; ++j)
      for (int r = 0; r < 8; ++r) {
        const int o = m0 + i * 16 + r + hi * 8;
        const int n = n0 + j * 16 + lrow;
        const size_t idx = (size_t)o * HW_DIM + n;
        Out[idx] = acc[i][j][r] + bo[o] + xres[idx];
      }
}

extern "C" void kernel_launch(void* const* d_in, const int* in_sizes, int n_in,
                              void* d_out, int out_size, void* d_ws, size_t ws_size,
                              hipStream_t stream) {
  const float* x     = (const float*)d_in[0];
  const float* Wq    = (const float*)d_in[1];
  const float* bq    = (const float*)d_in[2];
  const float* Wk    = (const float*)d_in[3];
  const float* bk    = (const float*)d_in[4];
  const float* Wv    = (const float*)d_in[5];
  const float* bv    = (const float*)d_in[6];
  const float* Wo    = (const float*)d_in[7];
  const float* bo    = (const float*)d_in[8];
  const float* gamma = (const float*)d_in[9];
  const float* beta  = (const float*)d_in[10];
  float* out = (float*)d_out;

  char* ws = (char*)d_ws;
  size_t off = 0;
  auto take = [&](size_t bytes) -> char* {
    char* p = ws + off;
    off += (bytes + 255) & ~(size_t)255;
    return p;
  };
  bf16_t* Wall = (bf16_t*)take((size_t)4 * C_DIM * C_DIM * 2);                 //  2 MB
  bf16_t* Xt   = (bf16_t*)take((size_t)B_DIM * HW_DIM * C_DIM * 2);           // 16 MB
  bf16_t* Qt   = (bf16_t*)take((size_t)GRP * HW_DIM * C_DIM * 2);             //  4 MB
  bf16_t* Kt   = (bf16_t*)take((size_t)GRP * HW_DIM * C_DIM * 2);             //  4 MB
  bf16_t* Vm   = (bf16_t*)take((size_t)GRP * HW_DIM * C_DIM * 2);             //  4 MB
  float*  S    = (float*)take((size_t)GRP * HW_DIM * HW_DIM * 4);             // 16 MB
  bf16_t* Attn = (bf16_t*)take((size_t)GRP * HW_DIM * HW_DIM * 2);            //  8 MB
  bf16_t* Hb   = (bf16_t*)take((size_t)GRP * HW_DIM * C_DIM * 2);             //  4 MB

  wconv_kernel<<<dim3(1024, 4), 256, 0, stream>>>(Wq, Wk, Wv, Wo, Wall);
  gn_silu_kernel<<<B_DIM * G_DIM, 256, 0, stream>>>(x, gamma, beta, Xt);

  const float qscale = 1.0f / sqrtf((float)C_DIM);
  for (int b0 = 0; b0 < B_DIM; b0 += GRP) {
    // 512x1024x512 GEMM x3 outputs x GRP batches
    qkv_gemm_kernel<<<dim3(32, 3, GRP), 256, 0, stream>>>(Wall, bq, bk, bv, Xt, b0,
                                                          Qt, Kt, Vm, qscale);
    // 1024x1024x512 -> scores
    scores_gemm_kernel<<<dim3(64, 1, GRP), 256, 0, stream>>>(Qt, Kt, S);
    softmax_kernel<<<dim3(HW_DIM, GRP), 256, 0, stream>>>(S, Attn);
    // 1024x512x1024 -> attn*V
    av_gemm_kernel<<<dim3(32, 1, GRP), 256, 0, stream>>>(Attn, Vm, Hb);
    // 512x1024x512 + bias + residual
    out_gemm_kernel<<<dim3(32, 1, GRP), 256, 0, stream>>>(Wall + (size_t)3 * C_DIM * C_DIM,
                                                          bo, Hb, x, out, b0);
  }
}
